// BaseAttention_1056561955023
// MI455X (gfx1250) — matrix operand order repaired and checked by reading, not compiled, not run on hardware
//
#include <hip/hip_runtime.h>
#include <hip/hip_bf16.h>

// ---------------------------------------------------------------------------
// Problem constants (from the reference)
// ---------------------------------------------------------------------------
#define BB 2
#define SS 2048
#define DD 1024
#define HH 16
#define KK 64
#define NN (HH * KK)   // 1024
#define BS (BB * SS)   // 4096

typedef __attribute__((ext_vector_type(16))) __bf16 v16bf;
typedef __attribute__((ext_vector_type(8)))  __bf16 bf16x8;
typedef __attribute__((ext_vector_type(8)))  float  v8f;

union Frag {
    v16bf  v;
    bf16x8 h[2];
};

#define WMMA_BF16(A, B, C) \
    __builtin_amdgcn_wmma_f32_16x16x32_bf16(false, (A), false, (B), (short)0, (C), false, false)

// ---------------------------------------------------------------------------
// Async global -> LDS copy (CDNA5 ASYNCcnt path).  Per ISA 10.2: generic
// pointers to LDS carry the LDS byte offset in addr[31:0], so truncation
// gives the dsaddr operand.  16 bytes per lane per issue.
// ---------------------------------------------------------------------------
__device__ __forceinline__ void async_load_b128(void* lds_ptr, const void* gaddr)
{
    unsigned lds_off = (unsigned)(unsigned long long)lds_ptr;
    asm volatile("global_load_async_to_lds_b128 %0, %1, off"
                 :: "v"(lds_off), "v"(gaddr)
                 : "memory");
}

__device__ __forceinline__ void wait_async0()
{
    asm volatile("s_wait_asynccnt 0x0" ::: "memory");
}

// ---------------------------------------------------------------------------
// Pre-pass: f32 -> bf16 convert (row-major copy)
// ---------------------------------------------------------------------------
__global__ __launch_bounds__(256)
void cvt_bf16_kernel(const float* __restrict__ in, __bf16* __restrict__ out, int n)
{
    int i = blockIdx.x * blockDim.x + threadIdx.x;
    if (i < n) out[i] = (__bf16)in[i];
}

// Pre-pass: transpose [rows][cols] f32 -> [cols][rows] bf16 (coalesced writes)
__global__ __launch_bounds__(256)
void transpose_bf16_kernel(const float* __restrict__ in, __bf16* __restrict__ out,
                           int rows, int cols)
{
    int i = blockIdx.x * blockDim.x + threadIdx.x;
    if (i < rows * cols) {
        int r = i % rows;
        int c = i / rows;
        out[(size_t)c * rows + r] = (__bf16)in[(size_t)r * cols + c];
    }
}

// ---------------------------------------------------------------------------
// bf16 GEMM: C[M][N] = A[M][Kd] * Bt^T  (Bt stored [N][Kd], both bf16)
// + bias[N], * scale.  128 threads = 4 waves; block tile 64x64.
// B tile (64 cols x 64 kd, 8 KB) staged in LDS via async loads and shared by
// all 4 waves; A fragments stream straight from global (per-wave rows).
// ---------------------------------------------------------------------------
template <bool F32OUT>
__global__ __launch_bounds__(128)
void gemm_bf16_kernel(const __bf16* __restrict__ A, const __bf16* __restrict__ Bt,
                      const float* __restrict__ bias, void* __restrict__ Cout,
                      int M, int N, int Kd, float scale)
{
    __shared__ __attribute__((aligned(32))) __bf16 btile[64 * 64];   // [col][kd]

    const int tid  = threadIdx.x;
    const int wave = tid >> 5;
    const int lane = tid & 31;
    const int lh   = lane >> 4;   // half-wave index (K-chunk selector)
    const int l15  = lane & 15;

    const int row0 = blockIdx.x * 64 + wave * 16;
    const int col0 = blockIdx.y * 64;

    v8f acc[4];
#pragma unroll
    for (int n = 0; n < 4; ++n)
#pragma unroll
        for (int j = 0; j < 8; ++j) acc[n][j] = 0.0f;

    const __bf16* arow = A + (size_t)(row0 + l15) * Kd;

    const int ccol = tid >> 1;      // cooperative-load column 0..63
    const int chal = tid & 1;       // 32-element half of the 64-deep k slab

    for (int kd = 0; kd < Kd; kd += 64) {
        // ---- async-stage Bt[col0..col0+63][kd..kd+63] into LDS ----
        {
            const __bf16* src = Bt + (size_t)(col0 + ccol) * Kd + kd + chal * 32;
            __bf16*       dst = btile + ccol * 64 + chal * 32;
            async_load_b128(dst,      src);
            async_load_b128(dst + 8,  src + 8);
            async_load_b128(dst + 16, src + 16);
            async_load_b128(dst + 24, src + 24);
        }
        wait_async0();
        __syncthreads();

#pragma unroll
        for (int ks = 0; ks < 2; ++ks) {
            // A fragment: row = l15; k = lh*16 + {0..15} (contig)
            Frag af;
            af.h[0] = *(const bf16x8*)(arow + kd + ks * 32 + lh * 16);
            af.h[1] = *(const bf16x8*)(arow + kd + ks * 32 + lh * 16 + 8);
#pragma unroll
            for (int n = 0; n < 4; ++n) {
                // B fragment: col = n*16 + l15; k = lh*16 + {0..15} (contig)
                Frag bf;
                bf.v = *(const v16bf*)(btile + (n * 16 + l15) * 64 + ks * 32 + lh * 16);
                acc[n] = WMMA_BF16(af.v, bf.v, acc[n]);
            }
        }
        __syncthreads();   // protect btile before next slab's overwrite
    }

#pragma unroll
    for (int n = 0; n < 4; ++n) {
        const int   col  = col0 + n * 16 + l15;
        const float bcol = bias ? bias[col] : 0.0f;
#pragma unroll
        for (int j = 0; j < 8; ++j) {
            const int    r   = row0 + lh * 8 + j;       // C layout: M = j + 8*lh
            const float  val = (acc[n][j] + bcol) * scale;
            const size_t idx = (size_t)r * N + col;
            if (F32OUT) ((float*)Cout)[idx] = val;
            else        ((__bf16*)Cout)[idx] = (__bf16)val;
        }
    }
}

// ---------------------------------------------------------------------------
// Flash-style attention per (b, h, 64-query tile).
// Q pre-scaled by 1/sqrt(K).  Q/K/V/ctx layout: [B][S][H][K] bf16.
// 128 threads = 4 waves; wave w handles query rows [qtile*64 + w*16, +16).
// K tile async-copied global->LDS; V transposed through registers.
// ---------------------------------------------------------------------------
__global__ __launch_bounds__(128)
void attn_kernel(const __bf16* __restrict__ Q, const __bf16* __restrict__ Kmat,
                 const __bf16* __restrict__ Vmat, __bf16* __restrict__ ctx)
{
    __shared__ __attribute__((aligned(32))) __bf16 kt[64 * 64];      // [key][kd]
    __shared__ __attribute__((aligned(32))) __bf16 vt[64 * 64];      // [n][key] (V^T)
    __shared__ __attribute__((aligned(32))) __bf16 ps[4][16 * 64];   // per-wave P stage

    const int qt   = blockIdx.x;
    const int h    = blockIdx.y;
    const int b    = blockIdx.z;
    const int tid  = threadIdx.x;
    const int wave = tid >> 5;
    const int lane = tid & 31;
    const int lh   = lane >> 4;
    const int l15  = lane & 15;

    const int qbase = qt * 64 + wave * 16;

    // Load this wave's Q fragments (16 rows x 64 kd) -> 2 k-step fragments
    Frag qf[2];
    {
        const size_t qrow = ((size_t)(b * SS + qbase + l15) * HH + h) * KK;
        qf[0].h[0] = *(const bf16x8*)(Q + qrow + lh * 16);
        qf[0].h[1] = *(const bf16x8*)(Q + qrow + lh * 16 + 8);
        qf[1].h[0] = *(const bf16x8*)(Q + qrow + 32 + lh * 16);
        qf[1].h[1] = *(const bf16x8*)(Q + qrow + 32 + lh * 16 + 8);
    }

    v8f oacc[4];
#pragma unroll
    for (int n = 0; n < 4; ++n)
#pragma unroll
        for (int j = 0; j < 8; ++j) oacc[n][j] = 0.0f;

    float rowm[8], rowl[8];
#pragma unroll
    for (int j = 0; j < 8; ++j) { rowm[j] = -1e30f; rowl[j] = 0.0f; }

    const int key  = tid >> 1;
    const int half = tid & 1;

    for (int kb = 0; kb < SS / 64; ++kb) {
        // ---- stage K (async, natural) and V (register transpose) tiles ----
        {
            const size_t grow = ((size_t)(b * SS + kb * 64 + key) * HH + h) * KK + half * 32;

            // K tile: contiguous 32 bf16 -> async global->LDS (ASYNCcnt path)
            const __bf16* ksrc = Kmat + grow;
            __bf16*       kdst = kt + key * 64 + half * 32;
            async_load_b128(kdst,      ksrc);
            async_load_b128(kdst + 8,  ksrc + 8);
            async_load_b128(kdst + 16, ksrc + 16);
            async_load_b128(kdst + 24, ksrc + 24);

            // V tile transposed: vt[n][key]
            Frag va, vb;
            va.v = *(const v16bf*)(Vmat + grow);
            vb.v = *(const v16bf*)(Vmat + grow + 16);
#pragma unroll
            for (int e = 0; e < 16; ++e) {
                vt[(half * 32 + e) * 64 + key]      = va.v[e];
                vt[(half * 32 + 16 + e) * 64 + key] = vb.v[e];
            }
        }
        wait_async0();
        __syncthreads();

        // ---- scores: S(16x64) = Q(16x64) * K^T ----
        v8f sc[4];
#pragma unroll
        for (int n = 0; n < 4; ++n) {
#pragma unroll
            for (int j = 0; j < 8; ++j) sc[n][j] = 0.0f;
#pragma unroll
            for (int ks = 0; ks < 2; ++ks) {
                Frag bf;   // B^T[n][kd] == K[key=n][kd] -> natural K rows
                bf.v = *(const v16bf*)(kt + (n * 16 + l15) * 64 + ks * 32 + lh * 16);
                sc[n] = WMMA_BF16(qf[ks].v, bf.v, sc[n]);
            }
        }

        // ---- online softmax update ----
#pragma unroll
        for (int j = 0; j < 8; ++j) {
            float mx = fmaxf(fmaxf(sc[0][j], sc[1][j]), fmaxf(sc[2][j], sc[3][j]));
#pragma unroll
            for (int m = 1; m < 16; m <<= 1) mx = fmaxf(mx, __shfl_xor(mx, m, 32));
            const float mnew  = fmaxf(rowm[j], mx);
            const float alpha = __expf(rowm[j] - mnew);
            rowm[j] = mnew;
            float rs = 0.0f;
#pragma unroll
            for (int n = 0; n < 4; ++n) {
                const float p = __expf(sc[n][j] - mnew);
                sc[n][j] = p;
                rs += p;
            }
#pragma unroll
            for (int m = 1; m < 16; m <<= 1) rs += __shfl_xor(rs, m, 32);
            rowl[j] = rowl[j] * alpha + rs;
#pragma unroll
            for (int n = 0; n < 4; ++n) oacc[n][j] *= alpha;
        }

        // ---- stage P (C/D layout -> LDS row-major) for A-layout reload ----
        __bf16* psw = ps[wave];
#pragma unroll
        for (int n = 0; n < 4; ++n)
#pragma unroll
            for (int j = 0; j < 8; ++j)
                psw[(lh * 8 + j) * 64 + n * 16 + l15] = (__bf16)sc[n][j];
        // same-wave LDS ops are in order; compiler inserts s_wait_dscnt

        Frag af[2];
#pragma unroll
        for (int ks = 0; ks < 2; ++ks) {
            af[ks].h[0] = *(const bf16x8*)(psw + l15 * 64 + ks * 32 + lh * 16);
            af[ks].h[1] = *(const bf16x8*)(psw + l15 * 64 + ks * 32 + lh * 16 + 8);
        }

        // ---- O += P * V ----
#pragma unroll
        for (int n = 0; n < 4; ++n) {
#pragma unroll
            for (int ks = 0; ks < 2; ++ks) {
                Frag bf;   // B^T[n][key] == vt rows (V already transposed)
                bf.v = *(const v16bf*)(vt + (n * 16 + l15) * 64 + ks * 32 + lh * 16);
                oacc[n] = WMMA_BF16(af[ks].v, bf.v, oacc[n]);
            }
        }
        __syncthreads();   // protect kt/vt before next block's overwrite
    }

    // ---- finalize: divide by softmax sum, store ctx bf16 ----
#pragma unroll
    for (int n = 0; n < 4; ++n)
#pragma unroll
        for (int j = 0; j < 8; ++j) {
            const int   srow = qbase + lh * 8 + j;
            const float o    = oacc[n][j] / rowl[j];
            ctx[((size_t)(b * SS + srow) * HH + h) * KK + n * 16 + l15] = (__bf16)o;
        }
}

// ---------------------------------------------------------------------------
// Residual + LayerNorm: out = LN(x + attn) * gamma + beta, eps = 1e-6
// One block (256 threads) per row of D=1024; 4 elements per thread.
// ---------------------------------------------------------------------------
__global__ __launch_bounds__(256)
void ln_kernel(const float* __restrict__ x, const float* __restrict__ attn,
               const float* __restrict__ gamma, const float* __restrict__ beta,
               float* __restrict__ out)
{
    __shared__ float s1[256];
    __shared__ float s2[256];

    const int row = blockIdx.x;
    const int tid = threadIdx.x;

    float y[4];
    float s = 0.0f, q = 0.0f;
#pragma unroll
    for (int e = 0; e < 4; ++e) {
        const int c = tid + e * 256;
        const float v = x[(size_t)row * DD + c] + attn[(size_t)row * DD + c];
        y[e] = v;
        s += v;
        q += v * v;
    }
    s1[tid] = s;
    s2[tid] = q;
    __syncthreads();
    for (int off = 128; off > 0; off >>= 1) {
        if (tid < off) { s1[tid] += s1[tid + off]; s2[tid] += s2[tid + off]; }
        __syncthreads();
    }
    const float mean = s1[0] * (1.0f / DD);
    const float var  = s2[0] * (1.0f / DD) - mean * mean;
    const float inv  = rsqrtf(var + 1e-6f);
#pragma unroll
    for (int e = 0; e < 4; ++e) {
        const int c = tid + e * 256;
        out[(size_t)row * DD + c] = (y[e] - mean) * inv * gamma[c] + beta[c];
    }
}

// ---------------------------------------------------------------------------
// Launch
// ---------------------------------------------------------------------------
extern "C" void kernel_launch(void* const* d_in, const int* in_sizes, int n_in,
                              void* d_out, int out_size, void* d_ws, size_t ws_size,
                              hipStream_t stream)
{
    (void)in_sizes; (void)n_in; (void)out_size; (void)ws_size;

    const float* x     = (const float*)d_in[0];
    const float* wq    = (const float*)d_in[1];
    const float* bq    = (const float*)d_in[2];
    const float* wk    = (const float*)d_in[3];
    const float* bk    = (const float*)d_in[4];
    const float* wv    = (const float*)d_in[5];
    const float* bv    = (const float*)d_in[6];
    const float* wo    = (const float*)d_in[7];
    const float* bo    = (const float*)d_in[8];
    const float* gamma = (const float*)d_in[9];
    const float* beta  = (const float*)d_in[10];

    // Workspace carve-up (~64 MB total)
    char*  p     = (char*)d_ws;
    auto   carve = [&](size_t bytes) {
        void* r = (void*)p;
        p += (bytes + 255) & ~(size_t)255;
        return r;
    };
    __bf16* xb   = (__bf16*)carve((size_t)BS * DD * 2);   // x in bf16
    __bf16* wqt  = (__bf16*)carve((size_t)DD * NN * 2);   // wq^T [N][D]
    __bf16* wkt  = (__bf16*)carve((size_t)DD * NN * 2);
    __bf16* wvt  = (__bf16*)carve((size_t)DD * NN * 2);
    __bf16* wot  = (__bf16*)carve((size_t)NN * DD * 2);   // wo^T [D][N]
    __bf16* Qb   = (__bf16*)carve((size_t)BS * NN * 2);   // Q (pre-scaled)
    __bf16* Kb   = (__bf16*)carve((size_t)BS * NN * 2);
    __bf16* Vb   = (__bf16*)carve((size_t)BS * NN * 2);
    __bf16* ctx  = (__bf16*)carve((size_t)BS * NN * 2);
    float*  attn = (float*) carve((size_t)BS * DD * 4);   // out-proj result

    // 1) pre-passes
    cvt_bf16_kernel<<<(BS * DD + 255) / 256, 256, 0, stream>>>(x, xb, BS * DD);
    transpose_bf16_kernel<<<(DD * NN + 255) / 256, 256, 0, stream>>>(wq, wqt, DD, NN);
    transpose_bf16_kernel<<<(DD * NN + 255) / 256, 256, 0, stream>>>(wk, wkt, DD, NN);
    transpose_bf16_kernel<<<(DD * NN + 255) / 256, 256, 0, stream>>>(wv, wvt, DD, NN);
    transpose_bf16_kernel<<<(NN * DD + 255) / 256, 256, 0, stream>>>(wo, wot, NN, DD);

    // 2) QKV projections (scale 1/sqrt(K)=0.125 folded into Q)
    dim3 gp(BS / 64, NN / 64);
    gemm_bf16_kernel<false><<<gp, 128, 0, stream>>>(xb, wqt, bq, (void*)Qb, BS, NN, DD, 0.125f);
    gemm_bf16_kernel<false><<<gp, 128, 0, stream>>>(xb, wkt, bk, (void*)Kb, BS, NN, DD, 1.0f);
    gemm_bf16_kernel<false><<<gp, 128, 0, stream>>>(xb, wvt, bv, (void*)Vb, BS, NN, DD, 1.0f);

    // 3) attention
    attn_kernel<<<dim3(SS / 64, HH, BB), 128, 0, stream>>>(Qb, Kb, Vb, ctx);

    // 4) output projection (f32 epilogue)
    dim3 go(BS / 64, DD / 64);
    gemm_bf16_kernel<true><<<go, 128, 0, stream>>>(ctx, wot, bo, (void*)attn, BS, DD, NN, 1.0f);

    // 5) residual + LayerNorm
    ln_kernel<<<BS, 256, 0, stream>>>(x, attn, gamma, beta, (float*)d_out);
}
